// DGCNN_14912126451984
// MI455X (gfx1250) — matrix operand order repaired and checked by reading, not compiled, not run on hardware
//
#include <hip/hip_runtime.h>
#include <hip/hip_bf16.h>

typedef __attribute__((ext_vector_type(16))) __bf16 v16bf;
typedef __attribute__((ext_vector_type(8)))  __bf16 v8bf;
typedef __attribute__((ext_vector_type(8)))  float  v8f;

#define KNN 20
#define B_  16
#define N_  4096
#define CT  8          // 16-wide column tiles per conv1 block

// ---------- helpers ----------
__device__ __forceinline__ __bf16 tobf(float f) {
    unsigned u = __float_as_uint(f);
    unsigned r = u + 0x7FFFu + ((u >> 16) & 1u);     // round-to-nearest-even
    return __builtin_bit_cast(__bf16, (unsigned short)(r >> 16));
}
// order-preserving float<->uint encoding (for atomicMax/Min on floats)
__device__ __forceinline__ unsigned encf(float f) {
    unsigned u = __float_as_uint(f);
    return (u & 0x80000000u) ? ~u : (u | 0x80000000u);
}
__device__ __forceinline__ float decf(unsigned e) {
    unsigned u = (e & 0x80000000u) ? (e & 0x7FFFFFFFu) : ~e;
    return __uint_as_float(u);
}
__device__ __forceinline__ v8f wmma_bf16(v16bf a, v16bf b, v8f c) {
    return __builtin_amdgcn_wmma_f32_16x16x32_bf16(false, a, false, b, (short)0, c,
                                                   false, false);
}
__device__ __forceinline__ float redsum16(float v) {
    v += __shfl_xor(v, 1, 32); v += __shfl_xor(v, 2, 32);
    v += __shfl_xor(v, 4, 32); v += __shfl_xor(v, 8, 32);
    return v;
}
__device__ __forceinline__ float redmax16(float v) {
    v = fmaxf(v, __shfl_xor(v, 1, 32)); v = fmaxf(v, __shfl_xor(v, 2, 32));
    v = fmaxf(v, __shfl_xor(v, 4, 32)); v = fmaxf(v, __shfl_xor(v, 8, 32));
    return v;
}
__device__ __forceinline__ float redmin16(float v) {
    v = fminf(v, __shfl_xor(v, 1, 32)); v = fminf(v, __shfl_xor(v, 2, 32));
    v = fminf(v, __shfl_xor(v, 4, 32)); v = fminf(v, __shfl_xor(v, 8, 32));
    return v;
}
// A fragment (16x32 bf16): row = matrix row for this lane (contiguous K storage).
__device__ __forceinline__ v16bf ldA16(const __bf16* __restrict__ row, int k0, int lane) {
    int h = (lane >> 4) << 3;
    v8bf lo = *(const v8bf*)(row + k0 + h);
    v8bf hi = *(const v8bf*)(row + k0 + 16 + h);
    return __builtin_shufflevector(lo, hi, 0, 1, 2, 3, 4, 5, 6, 7, 8, 9, 10, 11, 12,
                                   13, 14, 15);
}
// A fragment from LDS-staged weight tile
__device__ __forceinline__ v16bf ldA16s(const __bf16* row, int k0, int lane) {
    int h = (lane >> 4) << 3;
    v8bf lo = *(const v8bf*)(row + k0 + h);
    v8bf hi = *(const v8bf*)(row + k0 + 16 + h);
    return __builtin_shufflevector(lo, hi, 0, 1, 2, 3, 4, 5, 6, 7, 8, 9, 10, 11, 12,
                                   13, 14, 15);
}
// B fragment (32x16 bf16): row = this lane's column storage; K = 8h..8h+7, 16+8h..16+8h+7
__device__ __forceinline__ v16bf ldB16(const __bf16* __restrict__ row, int k0, int lane) {
    int kb = (lane >> 4) << 3;
    v8bf lo = *(const v8bf*)(row + k0 + kb);
    v8bf hi = *(const v8bf*)(row + k0 + 16 + kb);
    return __builtin_shufflevector(lo, hi, 0, 1, 2, 3, 4, 5, 6, 7, 8, 9, 10, 11, 12,
                                   13, 14, 15);
}

// ---------- prep kernels ----------
__global__ void wcvt_kernel(const float* __restrict__ W, int Cout, int Cin, int COUTP,
                            int CINP, __bf16* __restrict__ WB) {
    int i = blockIdx.x * blockDim.x + threadIdx.x;
    if (i >= COUTP * CINP) return;
    int o = i / CINP, c = i % CINP;
    float f = (o < Cout && c < Cin) ? W[(size_t)o * Cin + c] : 0.f;
    WB[i] = tobf(f);
}
__global__ void xcvt_kernel(const float* __restrict__ X, __bf16* __restrict__ XT) {
    int i = blockIdx.x * blockDim.x + threadIdx.x;
    if (i >= B_ * N_) return;
    int b = i / N_, n = i % N_;
    __bf16* row = XT + (size_t)i * 32;
#pragma unroll
    for (int c = 0; c < 32; ++c)
        row[c] = tobf(c < 3 ? X[((size_t)b * 3 + c) * N_ + n] : 0.f);
}
template <int CINP>
__global__ void sq_kernel(const __bf16* __restrict__ XT, float* __restrict__ SQ) {
    int i = blockIdx.x * blockDim.x + threadIdx.x;
    if (i >= B_ * N_) return;
    const unsigned* rw = (const unsigned*)(XT + (size_t)i * CINP);
    float s = 0.f;
#pragma unroll
    for (int t = 0; t < CINP / 2; ++t) {
        unsigned d = rw[t];
        float v0 = __uint_as_float(d << 16);
        float v1 = __uint_as_float(d & 0xFFFF0000u);
        s += v0 * v0 + v1 * v1;
    }
    SQ[i] = s;
}

// ---------- fused distance + top-k ----------
template <int CINP>
__global__ void __launch_bounds__(32)
knn_kernel(const __bf16* __restrict__ XT, const float* __restrict__ SQ,
           int* __restrict__ IDX) {
    int b = blockIdx.z, row0 = blockIdx.x * 32, lane = threadIdx.x;
    const __bf16* XTb = XT + (size_t)b * N_ * CINP;
    const float*  SQb = SQ + (size_t)b * N_;

    __shared__ float negd[32][17];
    __shared__ float rsq[32];
    rsq[lane] = SQb[row0 + lane];

    const __bf16* arow0 = XTb + (size_t)(row0 + (lane & 15)) * CINP;
    const __bf16* arow1 = arow0 + (size_t)16 * CINP;
    v16bf a00 = ldA16(arow0, 0, lane);
    v16bf a10 = ldA16(arow1, 0, lane);
    v16bf a01, a11;
    if (CINP > 32) { a01 = ldA16(arow0, 32, lane); a11 = ldA16(arow1, 32, lane); }

    float bd[KNN]; int bi[KNN];
#pragma unroll
    for (int j = 0; j < KNN; ++j) { bd[j] = -3.4e38f; bi[j] = 0; }

    for (int n0 = 0; n0 < N_; n0 += 16) {
        const __bf16* brow = XTb + (size_t)(n0 + (lane & 15)) * CINP;
        v16bf bf0 = ldB16(brow, 0, lane);
        v8f c0 = {}, c1 = {};
        c0 = wmma_bf16(a00, bf0, c0);
        c1 = wmma_bf16(a10, bf0, c1);
        if (CINP > 32) {
            v16bf bf1 = ldB16(brow, 32, lane);
            c0 = wmma_bf16(a01, bf1, c0);
            c1 = wmma_bf16(a11, bf1, c1);
        }
        int   colL = lane & 15;
        float csq  = SQb[n0 + colL];
        int   rb   = (lane >> 4) << 3;
        __syncthreads();
#pragma unroll
        for (int j = 0; j < 8; ++j) {
            negd[rb + j][colL]      = 2.f * c0[j] - rsq[rb + j]      - csq;
            negd[16 + rb + j][colL] = 2.f * c1[j] - rsq[16 + rb + j] - csq;
        }
        __syncthreads();
#pragma unroll
        for (int cc = 0; cc < 16; ++cc) {
            float d  = negd[lane][cc];
            int   ci = n0 + cc;
            if (d > bd[KNN - 1]) {
                float dd = d; int ii = ci;
#pragma unroll
                for (int j = 0; j < KNN; ++j) {
                    if (dd > bd[j]) {
                        float td = bd[j]; int ti = bi[j];
                        bd[j] = dd; bi[j] = ii; dd = td; ii = ti;
                    }
                }
            }
        }
    }
    size_t base = ((size_t)b * N_ + row0 + lane) * KNN;
#pragma unroll
    for (int j = 0; j < KNN; ++j) IDX[base + j] = bi[j];
}

// ---------- edge conv (gathered GEMM + per-(b,n,c) max/min over k + BN stats) ----------
template <int CINP>
__global__ void __launch_bounds__(32)
edgeconv_kernel(const __bf16* __restrict__ WB, const __bf16* __restrict__ XT,
                const int* __restrict__ IDX, float* __restrict__ sum,
                float* __restrict__ sumsq, unsigned* __restrict__ MXE,
                unsigned* __restrict__ MNE) {
    int b = blockIdx.z, m0 = blockIdx.y * 16, p0 = blockIdx.x * 16, lane = threadIdx.x;
    const __bf16* XTb = XT + (size_t)b * N_ * CINP;

    __shared__ unsigned smx[16][16], smn[16][16];   // [c_local][p_local]
    __shared__ int nidx[16 * KNN];
#pragma unroll
    for (int t = 0; t < 8; ++t) {
        int e = t * 32 + lane;
        ((unsigned*)smx)[e] = 0u;
        ((unsigned*)smn)[e] = 0xFFFFFFFFu;
    }
#pragma unroll
    for (int t = 0; t < 10; ++t) {
        int e = t * 32 + lane;
        nidx[e] = IDX[((size_t)b * N_ + p0 + e / KNN) * KNN + (e % KNN)];
    }
    __syncthreads();

    const __bf16* arow = WB + (size_t)(m0 + (lane & 15)) * CINP;
    v16bf w0 = ldA16(arow, 0, lane);
    v16bf w1;
    if (CINP > 32) w1 = ldA16(arow, 32, lane);

    float s8[8], q8[8];
#pragma unroll
    for (int j = 0; j < 8; ++j) { s8[j] = 0.f; q8[j] = 0.f; }

    for (int ct = 0; ct < KNN; ++ct) {
        int col  = ct * 16 + (lane & 15);
        int pl   = col / KNN, jn = col % KNN;
        int nsrc = nidx[pl * KNN + jn];
        const __bf16* brow = XTb + (size_t)nsrc * CINP;
        v8f acc = {};
        acc = wmma_bf16(w0, ldB16(brow, 0, lane), acc);
        if (CINP > 32) acc = wmma_bf16(w1, ldB16(brow, 32, lane), acc);
#pragma unroll
        for (int j = 0; j < 8; ++j) {
            float v = acc[j];
            s8[j] += v; q8[j] += v * v;
            int cl = j + ((lane >> 4) << 3);
            atomicMax(&smx[cl][pl], encf(v));
            atomicMin(&smn[cl][pl], encf(v));
        }
    }
    __syncthreads();
#pragma unroll
    for (int t = 0; t < 8; ++t) {
        int e = t * 32 + lane, cl = e >> 4, pl = e & 15;
        size_t o = ((size_t)b * N_ + p0 + pl) * 64 + m0 + cl;   // [B,N,64]
        MXE[o] = smx[cl][pl];
        MNE[o] = smn[cl][pl];
    }
#pragma unroll
    for (int j = 0; j < 8; ++j) {
        float s = redsum16(s8[j]);
        float q = redsum16(q8[j]);
        if ((lane & 15) == 0) {
            int c = m0 + j + ((lane >> 4) << 3);
            atomicAdd(&sum[c], s);
            atomicAdd(&sumsq[c], q);
        }
    }
}

// ---------- pointwise conv GEMM ----------
// Weight strip (16 x CIN bf16) is staged into LDS once per block with
// GLOBAL_LOAD_ASYNC_TO_LDS_B128 (ASYNCcnt), then A-fragments come from LDS.
template <int MODE, bool MINMAX, bool CONCAT, int COUT, int CIN, int CINX, int CR>
__global__ void __launch_bounds__(32)
conv1_kernel(const __bf16* __restrict__ WB, const __bf16* __restrict__ XT,
             const __bf16* __restrict__ X2, float* __restrict__ sum,
             float* __restrict__ sumsq, unsigned* __restrict__ MXE,
             unsigned* __restrict__ MNE, const float* __restrict__ scale,
             const float* __restrict__ bias, __bf16* __restrict__ Ybf,
             float* __restrict__ Yf) {
    int b = blockIdx.z, m0 = blockIdx.y * 16, nbase = blockIdx.x * 16 * CT,
        lane = threadIdx.x;
    int g = lane >> 4;
    const __bf16* XTb  = XT + (size_t)b * N_ * CINX;
    const __bf16* grow = CONCAT ? (X2 + (size_t)b * (CIN - CINX)) : nullptr;

    // ---- async-stage the 16 x CIN weight strip into LDS ----
    __shared__ alignas(32) __bf16 wtile[16 * CIN];
    {
        const char* wsrc  = (const char*)(WB + (size_t)m0 * CIN);
        unsigned    lbase = (unsigned)(size_t)(&wtile[0]);   // LDS offset = low 32 bits
        constexpr int TOTB = 16 * CIN * 2;                    // multiple of 512
#pragma unroll
        for (int t = 0; t < TOTB / (32 * 16); ++t) {
            int off = (t * 32 + lane) * 16;
            unsigned    laddr = lbase + off;
            const char* gaddr = wsrc + off;
            asm volatile("global_load_async_to_lds_b128 %0, %1, off"
                         :: "v"(laddr), "v"(gaddr)
                         : "memory");
        }
        asm volatile("s_wait_asynccnt 0x0" ::: "memory");
        __syncthreads();
    }
    const __bf16* arow = wtile + (size_t)(lane & 15) * CIN;

    float sc[8], bi[8];
    if (MODE == 1) {
#pragma unroll
        for (int j = 0; j < 8; ++j) {
            int c = m0 + j + (g << 3);
            sc[j] = scale[c]; bi[j] = bias[c];
        }
    }
    float s8[8], q8[8], mx8[8], mn8[8];
#pragma unroll
    for (int j = 0; j < 8; ++j) {
        s8[j] = 0.f; q8[j] = 0.f; mx8[j] = -3.4e38f; mn8[j] = 3.4e38f;
    }
    for (int ct = 0; ct < CT; ++ct) {
        int n0 = nbase + ct * 16;
        int n  = n0 + (lane & 15);
        const __bf16* xrow = XTb + (size_t)n * CINX;
        if (ct + 1 < CT)  // prefetch next activation row (global_prefetch_b8)
            __builtin_prefetch(xrow + (size_t)16 * CINX, 0, 0);
        v8f acc = {};
#pragma unroll
        for (int k0 = 0; k0 < CIN; k0 += 32) {
            v16bf a = ldA16s(arow, k0, lane);
            v16bf bb;
            if (CONCAT && k0 >= CINX) bb = ldB16(grow, k0 - CINX, lane);
            else                      bb = ldB16(xrow, k0, lane);
            acc = wmma_bf16(a, bb, acc);
        }
        if (MODE == 0) {
#pragma unroll
            for (int j = 0; j < 8; ++j) {
                float v = acc[j];
                s8[j] += v; q8[j] += v * v;
                if (MINMAX) { mx8[j] = fmaxf(mx8[j], v); mn8[j] = fminf(mn8[j], v); }
            }
        } else if (MODE == 1) {
            v8bf o;
#pragma unroll
            for (int j = 0; j < 8; ++j)
                o[j] = tobf(fmaxf(sc[j] * acc[j] + bi[j], 0.f));
            *(v8bf*)(Ybf + ((size_t)b * N_ + n) * COUT + m0 + (g << 3)) = o;
        } else {
#pragma unroll
            for (int j = 0; j < 8; ++j) {
                int c = m0 + j + (g << 3);
                if (c < CR)
                    Yf[((size_t)b * CR + c) * N_ + n] = fmaxf(acc[j], 0.f);
            }
        }
    }
    if (MODE == 0) {
#pragma unroll
        for (int j = 0; j < 8; ++j) {
            float s = redsum16(s8[j]);
            float q = redsum16(q8[j]);
            float mx = MINMAX ? redmax16(mx8[j]) : 0.f;
            float mn = MINMAX ? redmin16(mn8[j]) : 0.f;
            if ((lane & 15) == 0) {
                int c = m0 + j + (g << 3);
                atomicAdd(&sum[c], s);
                atomicAdd(&sumsq[c], q);
                if (MINMAX) {
                    atomicMax(&MXE[(size_t)b * COUT + c], encf(mx));
                    atomicMin(&MNE[(size_t)b * COUT + c], encf(mn));
                }
            }
        }
    }
}

// ---------- small epilogues ----------
__global__ void bnstats_kernel(const float* __restrict__ sum,
                               const float* __restrict__ sumsq,
                               const float* __restrict__ g,
                               const float* __restrict__ beta, float cnt, int C,
                               float* __restrict__ scale, float* __restrict__ bias) {
    int c = blockIdx.x * blockDim.x + threadIdx.x;
    if (c >= C) return;
    float m = sum[c] / cnt;
    float v = sumsq[c] / cnt - m * m;
    float a = g[c] * rsqrtf(fmaxf(v, 0.f) + 1e-5f);
    scale[c] = a;
    bias[c]  = beta[c] - m * a;
}
__global__ void edge_finalize_kernel(const unsigned* __restrict__ MXE,
                                     const unsigned* __restrict__ MNE,
                                     const float* __restrict__ scale,
                                     const float* __restrict__ bias,
                                     __bf16* __restrict__ OutXT) {
    int i = blockIdx.x * blockDim.x + threadIdx.x;
    if (i >= B_ * N_ * 64) return;
    int c = i & 63;
    float a = scale[c], bb = bias[c];
    float v = fmaxf(a * decf(MXE[i]) + bb, a * decf(MNE[i]) + bb);
    OutXT[i] = tobf(fmaxf(v, 0.f));
}
__global__ void glob_kernel(const unsigned* __restrict__ GMX,
                            const unsigned* __restrict__ GMN,
                            const float* __restrict__ scale,
                            const float* __restrict__ bias,
                            __bf16* __restrict__ GLOB) {
    int i = blockIdx.x * blockDim.x + threadIdx.x;
    if (i >= B_ * 1024) return;
    int c = i & 1023;
    float a = scale[c], bb = bias[c];
    GLOB[i] = tobf(fmaxf(fmaxf(a * decf(GMX[i]) + bb, a * decf(GMN[i]) + bb), 0.f));
}

// ---------- host orchestration ----------
template <int CINP>
static void run_edge_layer(const __bf16* XTin, const __bf16* WBl, const float* g,
                           const float* bt, __bf16* XTout, float* SQb, int* IDX,
                           float* SUM, float* SUMSQ, float* SCALE, float* BIAS,
                           unsigned* MXE, unsigned* MNE, hipStream_t stream) {
    dim3 w32(32);
    sq_kernel<CINP><<<(B_ * N_ + 255) / 256, 256, 0, stream>>>(XTin, SQb);
    knn_kernel<CINP><<<dim3(N_ / 32, 1, B_), w32, 0, stream>>>(XTin, SQb, IDX);
    hipMemsetAsync(SUM, 0, 64 * 4, stream);
    hipMemsetAsync(SUMSQ, 0, 64 * 4, stream);
    edgeconv_kernel<CINP><<<dim3(N_ / 16, 4, B_), w32, 0, stream>>>(
        WBl, XTin, IDX, SUM, SUMSQ, MXE, MNE);
    bnstats_kernel<<<1, 64, 0, stream>>>(SUM, SUMSQ, g, bt,
                                         (float)((size_t)B_ * KNN * N_), 64, SCALE,
                                         BIAS);
    edge_finalize_kernel<<<(B_ * N_ * 64 + 255) / 256, 256, 0, stream>>>(
        MXE, MNE, SCALE, BIAS, XTout);
}

extern "C" void kernel_launch(void* const* d_in, const int* in_sizes, int n_in,
                              void* d_out, int out_size, void* d_ws, size_t ws_size,
                              hipStream_t stream) {
    const float* x = (const float*)d_in[0];
    const float* Wf[9];
    for (int l = 1; l <= 8; ++l) Wf[l] = (const float*)d_in[1 + l];
    const float *G[8], *Bt[8];
    for (int l = 1; l <= 7; ++l) { G[l] = (const float*)d_in[10 + 2 * (l - 1)];
                                   Bt[l] = (const float*)d_in[11 + 2 * (l - 1)]; }

    size_t off = 0;
    auto alloc = [&](size_t bytes) -> void* {
        void* r = (char*)d_ws + off;
        off += (bytes + 255) & ~(size_t)255;
        return r;
    };
    __bf16* XT1  = (__bf16*)alloc((size_t)B_ * N_ * 32 * 2);
    __bf16* XTA  = (__bf16*)alloc((size_t)B_ * N_ * 64 * 2);
    __bf16* XTB  = (__bf16*)alloc((size_t)B_ * N_ * 64 * 2);
    __bf16* XTX1 = (__bf16*)alloc((size_t)B_ * N_ * 64 * 2);
    __bf16* XT6  = (__bf16*)alloc((size_t)B_ * N_ * 512 * 2);
    __bf16* XT7  = (__bf16*)alloc((size_t)B_ * N_ * 128 * 2);
    unsigned* MXE = (unsigned*)alloc((size_t)B_ * N_ * 64 * 4);
    unsigned* MNE = (unsigned*)alloc((size_t)B_ * N_ * 64 * 4);
    int*    IDX   = (int*)alloc((size_t)B_ * N_ * KNN * 4);
    float*  SQb   = (float*)alloc((size_t)B_ * N_ * 4);
    float*  SUM   = (float*)alloc(1024 * 4);
    float*  SUMSQ = (float*)alloc(1024 * 4);
    float*  SCALE = (float*)alloc(1024 * 4);
    float*  BIAS  = (float*)alloc(1024 * 4);
    unsigned* GMX = (unsigned*)alloc((size_t)B_ * 1024 * 4);
    unsigned* GMN = (unsigned*)alloc((size_t)B_ * 1024 * 4);
    __bf16* GLOBb = (__bf16*)alloc((size_t)B_ * 1024 * 2);
    __bf16* WB1 = (__bf16*)alloc(64 * 32 * 2);
    __bf16* WB2 = (__bf16*)alloc(64 * 64 * 2);
    __bf16* WB3 = (__bf16*)alloc(64 * 64 * 2);
    __bf16* WB4 = (__bf16*)alloc(64 * 64 * 2);
    __bf16* WB5 = (__bf16*)alloc(1024 * 64 * 2);
    __bf16* WB6 = (__bf16*)alloc(512 * 1088 * 2);
    __bf16* WB7 = (__bf16*)alloc(128 * 512 * 2);
    __bf16* WB8 = (__bf16*)alloc(64 * 128 * 2);
    (void)ws_size; (void)in_sizes; (void)n_in; (void)out_size;

    dim3 w32(32);
    auto wcvt = [&](const float* W, int Co, int Ci, int CoP, int CiP, __bf16* WB) {
        wcvt_kernel<<<(CoP * CiP + 255) / 256, 256, 0, stream>>>(W, Co, Ci, CoP, CiP,
                                                                 WB);
    };
    wcvt(Wf[1], 64, 3, 64, 32, WB1);
    wcvt(Wf[2], 64, 64, 64, 64, WB2);
    wcvt(Wf[3], 64, 64, 64, 64, WB3);
    wcvt(Wf[4], 64, 64, 64, 64, WB4);
    wcvt(Wf[5], 1024, 64, 1024, 64, WB5);
    wcvt(Wf[6], 512, 1088, 512, 1088, WB6);
    wcvt(Wf[7], 128, 512, 128, 512, WB7);
    wcvt(Wf[8], 50, 128, 64, 128, WB8);
    xcvt_kernel<<<(B_ * N_ + 255) / 256, 256, 0, stream>>>(x, XT1);

    run_edge_layer<32>(XT1, WB1, G[1], Bt[1], XTA, SQb, IDX, SUM, SUMSQ, SCALE, BIAS,
                       MXE, MNE, stream);                                   // h1
    run_edge_layer<64>(XTA, WB2, G[2], Bt[2], XTX1, SQb, IDX, SUM, SUMSQ, SCALE, BIAS,
                       MXE, MNE, stream);                                   // x1
    run_edge_layer<64>(XTX1, WB3, G[3], Bt[3], XTA, SQb, IDX, SUM, SUMSQ, SCALE, BIAS,
                       MXE, MNE, stream);                                   // h3
    run_edge_layer<64>(XTA, WB4, G[4], Bt[4], XTB, SQb, IDX, SUM, SUMSQ, SCALE, BIAS,
                       MXE, MNE, stream);                                   // h4

    const float cntN = (float)((size_t)B_ * N_);

    // 64 -> 1024: stats + per-(b,c) min/max only (full tensor never materialized)
    hipMemsetAsync(SUM, 0, 1024 * 4, stream);
    hipMemsetAsync(SUMSQ, 0, 1024 * 4, stream);
    hipMemsetAsync(GMX, 0x00, (size_t)B_ * 1024 * 4, stream);
    hipMemsetAsync(GMN, 0xFF, (size_t)B_ * 1024 * 4, stream);
    conv1_kernel<0, true, false, 1024, 64, 64, 1024>
        <<<dim3(N_ / (16 * CT), 1024 / 16, B_), w32, 0, stream>>>(
            WB5, XTB, nullptr, SUM, SUMSQ, GMX, GMN, nullptr, nullptr, nullptr,
            nullptr);
    bnstats_kernel<<<4, 256, 0, stream>>>(SUM, SUMSQ, G[5], Bt[5], cntN, 1024, SCALE,
                                          BIAS);
    glob_kernel<<<(B_ * 1024 + 255) / 256, 256, 0, stream>>>(GMX, GMN, SCALE, BIAS,
                                                             GLOBb);

    // 1088 -> 512 (concat X1 | broadcast glob): stats pass + apply pass (recompute)
    dim3 g6(N_ / (16 * CT), 512 / 16, B_);
    hipMemsetAsync(SUM, 0, 512 * 4, stream);
    hipMemsetAsync(SUMSQ, 0, 512 * 4, stream);
    conv1_kernel<0, false, true, 512, 1088, 64, 512><<<g6, w32, 0, stream>>>(
        WB6, XTX1, GLOBb, SUM, SUMSQ, nullptr, nullptr, nullptr, nullptr, nullptr,
        nullptr);
    bnstats_kernel<<<2, 256, 0, stream>>>(SUM, SUMSQ, G[6], Bt[6], cntN, 512, SCALE,
                                          BIAS);
    conv1_kernel<1, false, true, 512, 1088, 64, 512><<<g6, w32, 0, stream>>>(
        WB6, XTX1, GLOBb, nullptr, nullptr, nullptr, nullptr, SCALE, BIAS, XT6,
        nullptr);

    // 512 -> 128: stats + apply
    dim3 g7(N_ / (16 * CT), 128 / 16, B_);
    hipMemsetAsync(SUM, 0, 128 * 4, stream);
    hipMemsetAsync(SUMSQ, 0, 128 * 4, stream);
    conv1_kernel<0, false, false, 128, 512, 512, 128><<<g7, w32, 0, stream>>>(
        WB7, XT6, nullptr, SUM, SUMSQ, nullptr, nullptr, nullptr, nullptr, nullptr,
        nullptr);
    bnstats_kernel<<<1, 128, 0, stream>>>(SUM, SUMSQ, G[7], Bt[7], cntN, 128, SCALE,
                                          BIAS);
    conv1_kernel<1, false, false, 128, 512, 512, 128><<<g7, w32, 0, stream>>>(
        WB7, XT6, nullptr, nullptr, nullptr, nullptr, nullptr, SCALE, BIAS, XT7,
        nullptr);

    // 128 -> 50 + ReLU (no BN); rows padded to 64, store-guarded to 50
    conv1_kernel<2, false, false, 64, 128, 128, 50>
        <<<dim3(N_ / (16 * CT), 4, B_), w32, 0, stream>>>(
            WB8, XT7, nullptr, nullptr, nullptr, nullptr, nullptr, nullptr, nullptr,
            nullptr, (float*)d_out);
}
